// MultiLayerTimeAwareGNN_2748779069864
// MI455X (gfx1250) — matrix operand order repaired and checked by reading, not compiled, not run on hardware
//
#include <hip/hip_runtime.h>
#include <cstddef>
#include <cstdint>

typedef __attribute__((ext_vector_type(8)))  __bf16 v8bf;
typedef __attribute__((ext_vector_type(16))) __bf16 v16bf;
typedef __attribute__((ext_vector_type(8)))  float  v8f;

#define NN 20000
#define NE 320000
#define DM 256

// ---------------------------------------------------------------------------
// Pack a [Kreal, 256] f32 weight into WMMA B-fragment order (bf16).
// Output layout: ((ntile*ksteps + kstep)*32 + lane)*16 halves, where for lane
// (n = ntile*16 + lane%16, half = lane/16) element j*2+c holds
// B[kstep*32 + half*16 + 2j + c, n]   (zero-padded past Kreal).
// ---------------------------------------------------------------------------
__global__ void pack_weight_k(const float* __restrict__ W, __bf16* __restrict__ out,
                              int Kreal, int ksteps) {
  int tid = blockIdx.x * blockDim.x + threadIdx.x;
  int total = 16 * ksteps * 32;
  if (tid >= total) return;
  int lane  = tid & 31;
  int kstep = (tid >> 5) % ksteps;
  int l16 = lane & 15, half = lane >> 4;
  int ntile = tid / (32 * ksteps);
  int n = ntile * 16 + l16;
  __bf16* dst = out + (size_t)tid * 16;
#pragma unroll
  for (int j = 0; j < 8; ++j) {
#pragma unroll
    for (int c = 0; c < 2; ++c) {
      int k = kstep * 32 + half * 16 + 2 * j + c;
      float v = (k < Kreal) ? W[(size_t)k * DM + n] : 0.f;
      dst[j * 2 + c] = (__bf16)v;
    }
  }
}

// ---------------------------------------------------------------------------
// GEMM: C[M,256] = A[M,Kpad](bf16,row-major) x Bpacked + bias.
// One wave computes a 16x64 tile: 1 A-fragment feeds 4 WMMAs per k-step.
// ---------------------------------------------------------------------------
__global__ void gemm_bf16_wmma(const __bf16* __restrict__ A,
                               const __bf16* __restrict__ Bp,
                               const float* __restrict__ bias,
                               float* __restrict__ C,
                               int M, int Kpad, int hasBias) {
  int ksteps = Kpad >> 5;
  int wave = (blockIdx.x * blockDim.x + threadIdx.x) >> 5;
  int lane = threadIdx.x & 31;
  int mtiles = M >> 4;
  int ngroup = wave & 3;
  int mtile  = wave >> 2;
  if (mtile >= mtiles) return;
  int half = lane >> 4;
  int l16  = lane & 15;
  int m = (mtile << 4) + l16;
  const __bf16* arow = A + (size_t)m * Kpad;

  v8f acc[4] = {};
  for (int ks = 0; ks < ksteps; ++ks) {
    int kb = ks * 32 + half * 16;
    if (ks + 1 < ksteps)
      __builtin_prefetch(arow + kb + 32, 0, 1);       // global_prefetch_b8
    v8bf alo = *(const v8bf*)(arow + kb);
    v8bf ahi = *(const v8bf*)(arow + kb + 8);
    v16bf afrag = __builtin_shufflevector(alo, ahi,
        0,1,2,3,4,5,6,7,8,9,10,11,12,13,14,15);
#pragma unroll
    for (int s = 0; s < 4; ++s) {
      int ntile = ngroup * 4 + s;
      const __bf16* bptr = Bp + ((size_t)(ntile * ksteps + ks) * 32 + lane) * 16;
      v16bf bfrag = *(const v16bf*)bptr;
      acc[s] = __builtin_amdgcn_wmma_f32_16x16x32_bf16(
          false, afrag, false, bfrag, (short)0, acc[s], false, false);
    }
  }
#pragma unroll
  for (int s = 0; s < 4; ++s) {
    int col = ngroup * 64 + s * 16 + l16;
    float bv = hasBias ? bias[col] : 0.f;
#pragma unroll
    for (int r = 0; r < 8; ++r) {
      int row = (mtile << 4) + half * 8 + r;   // C/D layout: VGPR r <-> M=half*8+r
      C[(size_t)row * DM + col] = acc[s][r] + bv;
    }
  }
}

// ---------------------------------------------------------------------------
// Time2Vec + msg concat -> padded bf16 edge features [E, 96]
// ---------------------------------------------------------------------------
__global__ void edge_attr_k(const float* __restrict__ lu, const float* __restrict__ t,
                            const int* __restrict__ ei, const float* __restrict__ msg,
                            const float* __restrict__ wlin, const float* __restrict__ blin,
                            const float* __restrict__ w, const float* __restrict__ b,
                            __bf16* __restrict__ out) {
  int e = blockIdx.x * blockDim.x + threadIdx.x;
  if (e >= NE) return;
  int src = ei[e];
  float rt = lu[src] - t[e];
  __bf16* o = out + (size_t)e * 96;
  o[0] = (__bf16)(rt * wlin[0] + blin[0]);
#pragma unroll
  for (int j = 0; j < 8; ++j) o[1 + j] = (__bf16)sinf(rt * w[j] + b[j]);
  const float* m = msg + (size_t)e * 64;
  for (int j = 0; j < 64; ++j) o[9 + j] = (__bf16)m[j];
#pragma unroll
  for (int j = 73; j < 96; ++j) o[j] = (__bf16)0.f;
}

__global__ void pad_x_k(const float* __restrict__ x, __bf16* __restrict__ out) {
  size_t i = (size_t)blockIdx.x * blockDim.x + threadIdx.x;
  if (i >= (size_t)NN * 128) return;
  int n = (int)(i >> 7), c = (int)(i & 127);
  out[i] = (c < 100) ? (__bf16)x[(size_t)n * 100 + c] : (__bf16)0.f;
}

__global__ void f32_to_bf16_k(const float* __restrict__ in, __bf16* __restrict__ out, size_t n) {
  size_t i = (size_t)blockIdx.x * blockDim.x + threadIdx.x;
  if (i < n) out[i] = (__bf16)in[i];
}

// ---------------------------------------------------------------------------
// Pass 1: alpha[e,h] = scale * dot(q[dst], k[src]+e) per head; ordered-uint max
// ---------------------------------------------------------------------------
__device__ __forceinline__ unsigned f2key(float f) {
  unsigned b = __float_as_uint(f);
  return (b & 0x80000000u) ? ~b : (b | 0x80000000u);
}
__device__ __forceinline__ float key2f(unsigned k) {
  unsigned b = (k & 0x80000000u) ? (k & 0x7FFFFFFFu) : ~k;
  return __uint_as_float(b);
}

__global__ void edge_alpha_k(const float* __restrict__ q, const float* __restrict__ k,
                             const float* __restrict__ ebuf, const int* __restrict__ ei,
                             float* __restrict__ alpha, unsigned* __restrict__ nodemax) {
  int wid  = (blockIdx.x * blockDim.x + threadIdx.x) >> 5;
  int lane = threadIdx.x & 31;
  if (wid >= NE) return;
  int src = ei[wid], dst = ei[NE + wid];
  int head = lane >> 4;
  int base = head * 128 + (lane & 15) * 8;
  const float* qr = q + (size_t)dst * DM + base;
  const float* kr = k + (size_t)src * DM + base;
  const float* er = ebuf + (size_t)wid * DM + base;
  float s = 0.f;
#pragma unroll
  for (int c = 0; c < 8; ++c) s += qr[c] * (kr[c] + er[c]);
#pragma unroll
  for (int off = 1; off < 16; off <<= 1) s += __shfl_xor(s, off, 32);
  if ((lane & 15) == 0) {
    s *= 0.08838834764831845f;  // 1/sqrt(128)
    alpha[(size_t)wid * 2 + head] = s;
    atomicMax(nodemax + (size_t)dst * 2 + head, f2key(s));
  }
}

// Pass 2: a = exp(alpha - m[dst]); denom[dst] += a  (overwrite alpha with a)
__global__ void edge_exp_k(float* __restrict__ alpha, const unsigned* __restrict__ nodemax,
                           const int* __restrict__ ei, float* __restrict__ denom) {
  int i = blockIdx.x * blockDim.x + threadIdx.x;
  if (i >= NE * 2) return;
  int e = i >> 1, head = i & 1;
  int dst = ei[NE + e];
  float m = key2f(nodemax[(size_t)dst * 2 + head]);
  float a = __expf(alpha[i] - m);
  alpha[i] = a;
  atomicAdd(denom + (size_t)dst * 2 + head, a);
}

// Pass 3: aggr[dst] += (v[src]+e) * a/denom[dst]   (aggr pre-seeded with skip)
__global__ void edge_scatter_k(const float* __restrict__ v, const float* __restrict__ ebuf,
                               const float* __restrict__ alpha, const float* __restrict__ denom,
                               const int* __restrict__ ei, float* __restrict__ aggr) {
  int wid  = (blockIdx.x * blockDim.x + threadIdx.x) >> 5;
  int lane = threadIdx.x & 31;
  if (wid >= NE) return;
  int src = ei[wid], dst = ei[NE + wid];
  int head = lane >> 4;
  int base = head * 128 + (lane & 15) * 8;
  float w = alpha[(size_t)wid * 2 + head] / denom[(size_t)dst * 2 + head];
  const float* vr = v + (size_t)src * DM + base;
  const float* er = ebuf + (size_t)wid * DM + base;
  float* ar = aggr + (size_t)dst * DM + base;
#pragma unroll
  for (int c = 0; c < 8; ++c) atomicAdd(ar + c, (vr[c] + er[c]) * w);
}

// Post: h = silu(layernorm(h + aggr)) ; also emit bf16 copy for next GEMMs
__global__ void post_layer_k(float* __restrict__ h, const float* __restrict__ aggr,
                             const float* __restrict__ g, const float* __restrict__ b,
                             __bf16* __restrict__ hbf) {
  int node = (blockIdx.x * blockDim.x + threadIdx.x) >> 5;
  int lane = threadIdx.x & 31;
  if (node >= NN) return;
  int base = lane * 8;
  const float* hr = h + (size_t)node * DM;
  const float* ar = aggr + (size_t)node * DM;
  float y[8];
  float ps = 0.f;
#pragma unroll
  for (int c = 0; c < 8; ++c) { y[c] = hr[base + c] + ar[base + c]; ps += y[c]; }
#pragma unroll
  for (int off = 1; off < 32; off <<= 1) ps += __shfl_xor(ps, off, 32);
  float mu = ps * (1.f / 256.f);
  float pv = 0.f;
#pragma unroll
  for (int c = 0; c < 8; ++c) { float d = y[c] - mu; pv += d * d; }
#pragma unroll
  for (int off = 1; off < 32; off <<= 1) pv += __shfl_xor(pv, off, 32);
  float rstd = rsqrtf(pv * (1.f / 256.f) + 1e-5f);
#pragma unroll
  for (int c = 0; c < 8; ++c) {
    int col = base + c;
    float t = (y[c] - mu) * rstd * g[col] + b[col];
    float s = t / (1.f + __expf(-t));
    h[(size_t)node * DM + col] = s;
    hbf[(size_t)node * DM + col] = (__bf16)s;
  }
}

// ---------------------------------------------------------------------------
extern "C" void kernel_launch(void* const* d_in, const int* in_sizes, int n_in,
                              void* d_out, int out_size, void* d_ws, size_t ws_size,
                              hipStream_t stream) {
  const float* x    = (const float*)d_in[0];
  const float* lu   = (const float*)d_in[1];
  const int*   ei   = (const int*)  d_in[2];
  const float* t    = (const float*)d_in[3];
  const float* msg  = (const float*)d_in[4];
  const float* wlin = (const float*)d_in[5];
  const float* blin = (const float*)d_in[6];
  const float* t2w  = (const float*)d_in[7];
  const float* t2b  = (const float*)d_in[8];
  const float* linw = (const float*)d_in[9];
  const float* linb = (const float*)d_in[10];
  const float* Wq   = (const float*)d_in[11];
  const float* bq   = (const float*)d_in[12];
  const float* Wk   = (const float*)d_in[13];
  const float* bk   = (const float*)d_in[14];
  const float* Wv   = (const float*)d_in[15];
  const float* bv   = (const float*)d_in[16];
  const float* We   = (const float*)d_in[17];
  const float* Wsk  = (const float*)d_in[18];
  const float* bsk  = (const float*)d_in[19];
  const float* lng  = (const float*)d_in[20];
  const float* lnb  = (const float*)d_in[21];

  char* p = (char*)d_ws;
  auto alloc = [&](size_t bytes) -> char* {
    char* r = p; p += (bytes + 255) & ~(size_t)255; return r;
  };
  __bf16* hbf  = (__bf16*)alloc((size_t)NN * DM * 2);
  __bf16* xbf  = (__bf16*)alloc((size_t)NN * 128 * 2);
  __bf16* eat  = (__bf16*)alloc((size_t)NE * 96 * 2);
  float*  h    = (float*) alloc((size_t)NN * DM * 4);
  float*  q    = (float*) alloc((size_t)NN * DM * 4);
  float*  k    = (float*) alloc((size_t)NN * DM * 4);
  float*  v    = (float*) alloc((size_t)NN * DM * 4);
  float*  aggr = (float*) alloc((size_t)NN * DM * 4);
  float*  ebuf = (float*) alloc((size_t)NE * DM * 4);
  float*  alp  = (float*) alloc((size_t)NE * 2 * 4);
  unsigned* nmax = (unsigned*)alloc((size_t)NN * 2 * 4);
  float*  den  = (float*) alloc((size_t)NN * 2 * 4);
  __bf16* linp = (__bf16*)alloc((size_t)16 * 4 * 32 * 16 * 2);
  __bf16* Wqp  = (__bf16*)alloc((size_t)3 * 16 * 8 * 32 * 16 * 2);
  __bf16* Wkp  = (__bf16*)alloc((size_t)3 * 16 * 8 * 32 * 16 * 2);
  __bf16* Wvp  = (__bf16*)alloc((size_t)3 * 16 * 8 * 32 * 16 * 2);
  __bf16* Wsp  = (__bf16*)alloc((size_t)3 * 16 * 8 * 32 * 16 * 2);
  __bf16* Wep  = (__bf16*)alloc((size_t)3 * 16 * 3 * 32 * 16 * 2);

  auto gemm = [&](const __bf16* A, const __bf16* Bp, const float* bias,
                  float* C, int M, int Kpad) {
    int waves = (M / 16) * 4;
    int blocks = (waves * 32 + 255) / 256;
    gemm_bf16_wmma<<<blocks, 256, 0, stream>>>(A, Bp, bias, C, M, Kpad,
                                               bias != nullptr);
  };

  // -------- weight packing (fragment order, bf16) --------
  pack_weight_k<<<(16*4*32 + 255)/256, 256, 0, stream>>>(linw, linp, 100, 4);
  for (int l = 0; l < 3; ++l) {
    size_t wo = (size_t)l * 256 * 256;          // f32 weight offset
    size_t po = (size_t)l * 16 * 8 * 32 * 16;   // packed halves offset
    pack_weight_k<<<(16*8*32 + 255)/256, 256, 0, stream>>>(Wq  + wo, Wqp + po, 256, 8);
    pack_weight_k<<<(16*8*32 + 255)/256, 256, 0, stream>>>(Wk  + wo, Wkp + po, 256, 8);
    pack_weight_k<<<(16*8*32 + 255)/256, 256, 0, stream>>>(Wv  + wo, Wvp + po, 256, 8);
    pack_weight_k<<<(16*8*32 + 255)/256, 256, 0, stream>>>(Wsk + wo, Wsp + po, 256, 8);
    pack_weight_k<<<(16*3*32 + 255)/256, 256, 0, stream>>>(We + (size_t)l*73*256,
                                                           Wep + (size_t)l*16*3*32*16, 73, 3);
  }

  // -------- input prep --------
  pad_x_k<<<((size_t)NN*128 + 255)/256, 256, 0, stream>>>(x, xbf);
  edge_attr_k<<<(NE + 255)/256, 256, 0, stream>>>(lu, t, ei, msg, wlin, blin,
                                                  t2w, t2b, eat);

  // -------- h0 = x @ lin_w + lin_b --------
  gemm(xbf, linp, linb, h, NN, 128);
  f32_to_bf16_k<<<((size_t)NN*DM + 255)/256, 256, 0, stream>>>(h, hbf, (size_t)NN*DM);

  // -------- 3 attention layers --------
  for (int l = 0; l < 3; ++l) {
    size_t po = (size_t)l * 16 * 8 * 32 * 16;
    gemm(hbf, Wqp + po, bq  + l*DM, q,    NN, 256);
    gemm(hbf, Wkp + po, bk  + l*DM, k,    NN, 256);
    gemm(hbf, Wvp + po, bv  + l*DM, v,    NN, 256);
    gemm(hbf, Wsp + po, bsk + l*DM, aggr, NN, 256);   // skip seeds aggr
    gemm(eat, Wep + (size_t)l*16*3*32*16, nullptr, ebuf, NE, 96);

    hipMemsetAsync(nmax, 0, (size_t)NN * 2 * 4, stream);
    hipMemsetAsync(den,  0, (size_t)NN * 2 * 4, stream);

    edge_alpha_k<<<(NE*32 + 255)/256, 256, 0, stream>>>(q, k, ebuf, ei, alp, nmax);
    edge_exp_k<<<(NE*2 + 255)/256, 256, 0, stream>>>(alp, nmax, ei, den);
    edge_scatter_k<<<(NE*32 + 255)/256, 256, 0, stream>>>(v, ebuf, alp, den, ei, aggr);
    post_layer_k<<<(NN*32 + 255)/256, 256, 0, stream>>>(h, aggr, lng + l*DM,
                                                        lnb + l*DM, hbf);
  }

  hipMemcpyAsync(d_out, h, (size_t)NN * DM * sizeof(float),
                 hipMemcpyDeviceToDevice, stream);
}
